// SuperPoint_31636729103215
// MI455X (gfx1250) — matrix operand order repaired and checked by reading, not compiled, not run on hardware
//
#include <hip/hip_runtime.h>

typedef __attribute__((ext_vector_type(16))) _Float16 v16h;
typedef __attribute__((ext_vector_type(8)))  _Float16 v8h;
typedef __attribute__((ext_vector_type(8)))  float    v8f;

__device__ __forceinline__ int iclampi(int v, int lo, int hi) {
  return v < lo ? lo : (v > hi ? hi : v);
}

// ---------------------------------------------------------------------------
// Weight packing into WMMA A-fragment layout.
// A 16x32 fp16 fragment (CDNA5 ISA): lane l holds row M=l&15;
// halves h=0..15 are packed as K = (l>>4)*16 + h  (per 32-K chunk, same order as B).
// packed[((ct*NK + kc)*32 + l)*16 + h]
// ---------------------------------------------------------------------------
__global__ void k_pack_w(const float* __restrict__ w, _Float16* __restrict__ out,
                         int CIN, int KHW, int NK, int total, int Oreal) {
  int t = blockIdx.x * 256 + threadIdx.x;
  if (t >= total) return;
  int h  = t & 15;
  int l  = (t >> 4) & 31;
  int kc = (t >> 9) % NK;
  int ct = t / (512 * NK);
  int laneHi = l >> 4;
  int K = kc * 32 + laneHi * 16 + h;
  int o = ct * 16 + (l & 15);
  float val = 0.f;
  if (o < Oreal) {
    if (KHW == 9) {
      int c = K % CIN;
      int kyx = K / CIN;
      val = w[((size_t)o * CIN + c) * 9 + kyx];
    } else {
      val = w[(size_t)o * CIN + K];
    }
  }
  out[t] = (_Float16)val;
}

// ---------------------------------------------------------------------------
// conv1a: Cin=1, memory bound -> plain VALU kernel, fp32 img -> fp16 NHWC(64)
// ---------------------------------------------------------------------------
__global__ __launch_bounds__(256)
void k_conv1a(const float* __restrict__ img, const float* __restrict__ w,
              const float* __restrict__ bias, _Float16* __restrict__ out) {
  size_t t = (size_t)blockIdx.x * 256 + threadIdx.x;
  const size_t total = (size_t)4 * 480 * 640 * 64;
  if (t >= total) return;
  int o = (int)(t & 63);
  size_t pix = t >> 6;
  int x = (int)(pix % 640);
  size_t r = pix / 640;
  int y = (int)(r % 480);
  int b = (int)(r / 480);
  float acc = bias[o];
  const float* ib = img + (size_t)b * 480 * 640;
  for (int ky = 0; ky < 3; ++ky) {
    int gy = y + ky - 1;
    if (gy < 0 || gy >= 480) continue;
    for (int kx = 0; kx < 3; ++kx) {
      int gx = x + kx - 1;
      if (gx < 0 || gx >= 640) continue;
      acc += ib[(size_t)gy * 640 + gx] * w[o * 9 + ky * 3 + kx];
    }
  }
  out[t] = (_Float16)fmaxf(acc, 0.f);
}

// ---------------------------------------------------------------------------
// WMMA implicit-GEMM conv3x3, fp16 NHWC in/out, bias+ReLU fused.
// Block = 256 threads = 8 waves. Wave (mi,ni) computes 16 couts x 16 pixels.
// NSUB = pixel subtiles per block (1 or 2). LDS stages 3 x (TILE+2) x CIN halo.
// ---------------------------------------------------------------------------
template <int CIN, int COUT, int NSUB, bool RELU>
__global__ __launch_bounds__(256)
void k_conv3x3_wmma(const _Float16* __restrict__ in,
                    const _Float16* __restrict__ wpk,
                    const float* __restrict__ bias,
                    _Float16* __restrict__ out, int H, int W) {
  constexpr int TILE = NSUB * 16;
  constexpr int MSUB = 8 / NSUB;
  constexpr int CPB  = MSUB * 16;           // couts per block
  constexpr int COLS = TILE + 2;
  constexpr int NK   = CIN * 9 / 32;        // K chunks of 32
  __shared__ alignas(32) _Float16 lds[3 * COLS * CIN];

  const int tid = threadIdx.x;
  const int xt  = blockIdx.x;
  const int y   = blockIdx.y;
  const int zc  = blockIdx.z;
  constexpr int CTILES = COUT / CPB;
  const int b   = zc / CTILES;
  const int ctb = zc % CTILES;
  const int coutBase = ctb * CPB;

  // ---- stage input halo tile (zero-padded) as 32-bit pieces ----
  const int totalPairs = 3 * COLS * CIN / 2;
  for (int i = tid; i < totalPairs; i += 256) {
    int e = i * 2;
    int ch = e % CIN;
    int rest = e / CIN;
    int col = rest % COLS;
    int row = rest / COLS;
    int gy = y + row - 1;
    int gx = xt * TILE + col - 1;
    unsigned v = 0u;
    if (gy >= 0 && gy < H && gx >= 0 && gx < W)
      v = *(const unsigned*)(in + ((size_t)(b * H + gy) * W + gx) * CIN + ch);
    *(unsigned*)(lds + e) = v;
  }
  __syncthreads();

  const int wv = tid >> 5, lane = tid & 31;
  const int mi = wv / NSUB, ni = wv % NSUB;
  const int laneHi = lane >> 4, lanLo = lane & 15;
  const int ct = coutBase / 16 + mi;

  v8f acc = {};
  const _Float16* aptr = wpk + ((size_t)ct * NK * 32 + lane) * 16;
  for (int kc = 0; kc < NK; ++kc) {
    v16h A = *(const v16h*)aptr;
    aptr += 512;
    __builtin_prefetch((const void*)aptr, 0, 0);
    int kbase = kc * 32;
    int kyx = kbase / CIN;
    int cBase = kbase % CIN;
    int ky = kyx / 3, kx = kyx % 3;
    int col = ni * 16 + lanLo + kx;
    // B 32x16 fragment: lane l = column n, halves h -> K = (l>>4)*16 + h
    v16h Bf = *(const v16h*)(lds + ((size_t)(ky * COLS + col) * CIN + cBase + laneHi * 16));
    acc = __builtin_amdgcn_wmma_f32_16x16x32_f16(false, A, false, Bf,
                                                 (short)0, acc, false, false);
  }

  // epilogue: lane holds couts [laneHi*8, laneHi*8+8) for pixel n=lanLo
  int x = xt * TILE + ni * 16 + lanLo;
  int coutLo = coutBase + mi * 16 + laneHi * 8;
  v8h hv;
  for (int v = 0; v < 8; ++v) {
    float val = acc[v] + bias[coutLo + v];
    if (RELU) val = fmaxf(val, 0.f);
    hv[v] = (_Float16)val;
  }
  *(v8h*)(out + ((size_t)(b * H + y) * W + x) * COUT + coutLo) = hv;
}

// ---------------------------------------------------------------------------
// WMMA 1x1 conv (GEMM over pixels), fp16 in, fp32 out (heads).
// ---------------------------------------------------------------------------
template <int CIN, int COUTP, bool RELU>
__global__ __launch_bounds__(256)
void k_conv1x1_wmma(const _Float16* __restrict__ in,
                    const _Float16* __restrict__ wpk,
                    const float* __restrict__ bias, int Oreal,
                    float* __restrict__ out, int HW) {
  constexpr int NK = CIN / 32;
  const int tid = threadIdx.x;
  const int wv = tid >> 5, lane = tid & 31;
  const int laneHi = lane >> 4, lanLo = lane & 15;
  const int b = blockIdx.y;
  const int miG = blockIdx.z * 8 + wv;
  if (miG * 16 >= COUTP) return;          // wave-uniform skip (EXEC stays full)
  const int pix = blockIdx.x * 16 + lanLo;
  const _Float16* bp = in + ((size_t)b * HW + pix) * CIN + laneHi * 16;
  const _Float16* aptr = wpk + ((size_t)miG * NK * 32 + lane) * 16;
  v8f acc = {};
  for (int kc = 0; kc < NK; ++kc) {
    v16h A = *(const v16h*)aptr;
    aptr += 512;
    v16h Bf = *(const v16h*)(bp + kc * 32);
    acc = __builtin_amdgcn_wmma_f32_16x16x32_f16(false, A, false, Bf,
                                                 (short)0, acc, false, false);
  }
  int coutLo = miG * 16 + laneHi * 8;
  float vals[8];
  for (int v = 0; v < 8; ++v) {
    int co = coutLo + v;
    float bb = (co < Oreal) ? bias[co] : 0.f;
    float val = acc[v] + bb;
    if (RELU) val = fmaxf(val, 0.f);
    vals[v] = val;
  }
  float* o = out + ((size_t)b * HW + pix) * COUTP + coutLo;
  *(float4*)(o)     = make_float4(vals[0], vals[1], vals[2], vals[3]);
  *(float4*)(o + 4) = make_float4(vals[4], vals[5], vals[6], vals[7]);
}

// ---------------------------------------------------------------------------
// 2x2 max pool, fp16 NHWC
// ---------------------------------------------------------------------------
__global__ __launch_bounds__(256)
void k_maxpool2(const _Float16* __restrict__ in, _Float16* __restrict__ out,
                int Ho, int Wo, int C) {
  size_t total = (size_t)4 * Ho * Wo * C;
  size_t t = (size_t)blockIdx.x * 256 + threadIdx.x;
  if (t >= total) return;
  int c = (int)(t % C);
  size_t r = t / C;
  int wo = (int)(r % Wo); r /= Wo;
  int ho = (int)(r % Ho);
  int b  = (int)(r / Ho);
  int Wi = Wo * 2, Hi = Ho * 2;
  const _Float16* p = in + (((size_t)(b * Hi + ho * 2) * Wi + wo * 2) * C + c);
  size_t rs = (size_t)Wi * C;
  float m = fmaxf(fmaxf((float)p[0], (float)p[C]),
                  fmaxf((float)p[rs], (float)p[rs + C]));
  out[t] = (_Float16)m;
}

// ---------------------------------------------------------------------------
// softmax(65) + drop last + pixel shuffle 8x8 -> dense scores [4][480][640]
// ---------------------------------------------------------------------------
__global__ __launch_bounds__(256)
void k_softmax_shuffle(const float* __restrict__ pb, float* __restrict__ s0) {
  int t = blockIdx.x * 256 + threadIdx.x;   // over 4*60*80 cells
  if (t >= 4 * 60 * 80) return;
  int xc = t % 80;
  int r = t / 80;
  int yc = r % 60;
  int b = r / 60;
  const float* v = pb + (size_t)t * 80;     // stride-80 storage, 65 valid
  float mx = v[0];
  for (int i = 1; i < 65; ++i) mx = fmaxf(mx, v[i]);
  float sum = 0.f;
  for (int i = 0; i < 65; ++i) sum += expf(v[i] - mx);
  float inv = 1.f / sum;
  float* ob = s0 + (size_t)b * 480 * 640;
  for (int cy = 0; cy < 8; ++cy)
    for (int cx = 0; cx < 8; ++cx)
      ob[(size_t)(yc * 8 + cy) * 640 + (xc * 8 + cx)] =
          expf(v[cy * 8 + cx] - mx) * inv;
}

// ---------------------------------------------------------------------------
// separable 9x9 max pool over [4][480][640] (clipped window == -inf pad)
// ---------------------------------------------------------------------------
__global__ __launch_bounds__(256)
void k_pool9h(const float* __restrict__ in, float* __restrict__ out) {
  size_t t = (size_t)blockIdx.x * 256 + threadIdx.x;
  if (t >= (size_t)4 * 480 * 640) return;
  int x = (int)(t % 640);
  float m = -3.4e38f;
  for (int dx = -4; dx <= 4; ++dx) {
    int xx = x + dx;
    if (xx >= 0 && xx < 640) m = fmaxf(m, in[t - x + xx]);
  }
  out[t] = m;
}

__global__ __launch_bounds__(256)
void k_pool9v(const float* __restrict__ in, float* __restrict__ out) {
  size_t t = (size_t)blockIdx.x * 256 + threadIdx.x;
  if (t >= (size_t)4 * 480 * 640) return;
  int y = (int)((t / 640) % 480);
  float m = -3.4e38f;
  for (int dy = -4; dy <= 4; ++dy) {
    int yy = y + dy;
    if (yy >= 0 && yy < 480) m = fmaxf(m, in[t + (size_t)(dy) * 640]);
  }
  out[t] = m;
}

__global__ __launch_bounds__(256)
void k_eqmask(const float* __restrict__ s, const float* __restrict__ p,
              float* __restrict__ m) {
  size_t t = (size_t)blockIdx.x * 256 + threadIdx.x;
  if (t >= (size_t)4 * 480 * 640) return;
  m[t] = (s[t] == p[t]) ? 1.f : 0.f;
}

__global__ __launch_bounds__(256)
void k_suppress(const float* __restrict__ s, const float* __restrict__ p,
                float* __restrict__ ss) {
  size_t t = (size_t)blockIdx.x * 256 + threadIdx.x;
  if (t >= (size_t)4 * 480 * 640) return;
  ss[t] = (p[t] > 0.f) ? 0.f : s[t];
}

__global__ __launch_bounds__(256)
void k_update(float* __restrict__ m, const float* __restrict__ ss,
              const float* __restrict__ p2, const float* __restrict__ p) {
  size_t t = (size_t)blockIdx.x * 256 + threadIdx.x;
  if (t >= (size_t)4 * 480 * 640) return;
  bool nm = (ss[t] == p2[t]) && (p[t] <= 0.f);
  m[t] = (m[t] != 0.f || nm) ? 1.f : 0.f;
}

// ---------------------------------------------------------------------------
// top-k: compact survivors then per-batch bitonic sort of 8192 keys in LDS
// ---------------------------------------------------------------------------
__global__ void k_zero4(int* c) { if (threadIdx.x < 4) c[threadIdx.x] = 0; }

__global__ __launch_bounds__(256)
void k_compact(const float* __restrict__ s0, const float* __restrict__ m,
               float* __restrict__ cScore, int* __restrict__ cIdx,
               int* __restrict__ cnt) {
  size_t t = (size_t)blockIdx.x * 256 + threadIdx.x;
  if (t >= (size_t)4 * 307200) return;
  int b = (int)(t / 307200);
  int pix = (int)(t % 307200);
  int x = pix % 640, y = pix / 640;
  if (m[t] != 0.f && x >= 4 && x < 636 && y >= 4 && y < 476) {
    int slot = atomicAdd(&cnt[b], 1);
    if (slot < 8192) {
      cScore[b * 8192 + slot] = s0[t];
      cIdx[b * 8192 + slot] = pix;
    }
  }
}

__global__ __launch_bounds__(1024)
void k_topk_sort(const float* __restrict__ cScore, const int* __restrict__ cIdx,
                 const int* __restrict__ cnt, float* __restrict__ dout) {
  __shared__ unsigned long long keys[8192];
  const int b = blockIdx.x, tid = threadIdx.x;
  int n = cnt[b];
  if (n > 8192) n = 8192;
  for (int i = tid; i < 8192; i += 1024) {
    unsigned long long key = 0ull;
    if (i < n) {
      unsigned sb = __float_as_uint(cScore[b * 8192 + i]);
      unsigned inv = 0x7FFFFFFFu - (unsigned)cIdx[b * 8192 + i];
      key = ((unsigned long long)sb << 32) | (unsigned long long)inv;
    }
    keys[i] = key;
  }
  __syncthreads();
  for (int kk = 2; kk <= 8192; kk <<= 1) {
    for (int j = kk >> 1; j > 0; j >>= 1) {
      for (int i = tid; i < 8192; i += 1024) {
        int ixj = i ^ j;
        if (ixj > i) {
          unsigned long long a = keys[i], c = keys[ixj];
          bool up = (i & kk) == 0;           // descending overall
          bool sw = up ? (a < c) : (a > c);
          if (sw) { keys[i] = c; keys[ixj] = a; }
        }
      }
      __syncthreads();
    }
  }
  if (tid < 1024) {
    unsigned long long key = keys[tid];
    float sc = __uint_as_float((unsigned)(key >> 32));
    unsigned pix = 0x7FFFFFFFu - (unsigned)(key & 0xFFFFFFFFu);
    float kx = 0.f, ky = 0.f, val = 0.f;
    if (key != 0ull) {
      kx = (float)(pix % 640);
      ky = (float)(pix / 640);
      val = sc;
    }
    dout[(size_t)(b * 1024 + tid) * 2 + 0] = kx;
    dout[(size_t)(b * 1024 + tid) * 2 + 1] = ky;
    dout[(size_t)4 * 1024 * 2 + b * 1024 + tid] = (val > 0.005f) ? val : 0.f;
  }
}

// ---------------------------------------------------------------------------
// descriptor: per-pixel L2 norm (in place), bilinear sample + final L2 norm
// ---------------------------------------------------------------------------
__global__ __launch_bounds__(256)
void k_desc_norm(float* __restrict__ d) {
  const int pix = blockIdx.x;     // 4*4800
  const int c = threadIdx.x;
  __shared__ float red[256];
  float v = d[(size_t)pix * 256 + c];
  red[c] = v * v;
  __syncthreads();
  for (int s = 128; s > 0; s >>= 1) {
    if (c < s) red[c] += red[c + s];
    __syncthreads();
  }
  float nrm = fmaxf(sqrtf(red[0]), 1e-12f);
  d[(size_t)pix * 256 + c] = v / nrm;
}

__global__ __launch_bounds__(256)
void k_desc_sample(const float* __restrict__ d, const float* __restrict__ kpts,
                   float* __restrict__ desc_out) {
  const int blk = blockIdx.x;     // 4*1024
  const int b = blk >> 10, t = blk & 1023;
  const int c = threadIdx.x;
  float kx = kpts[(size_t)(b * 1024 + t) * 2 + 0];
  float ky = kpts[(size_t)(b * 1024 + t) * 2 + 1];
  // kn*(w-1): (kx - 3.5)/(640 - 4.5) * 79,  (ky - 3.5)/(480 - 4.5) * 59
  float ix = (kx - 3.5f) * (79.0f / 635.5f);
  float iy = (ky - 3.5f) * (59.0f / 475.5f);
  float x0f = floorf(ix), y0f = floorf(iy);
  float wx = ix - x0f, wy = iy - y0f;
  int x0 = iclampi((int)x0f, 0, 79), x1 = iclampi((int)x0f + 1, 0, 79);
  int y0 = iclampi((int)y0f, 0, 59), y1 = iclampi((int)y0f + 1, 0, 59);
  const float* base = d + (size_t)b * 4800 * 256;
  float d00 = base[(size_t)(y0 * 80 + x0) * 256 + c];
  float d01 = base[(size_t)(y0 * 80 + x1) * 256 + c];
  float d10 = base[(size_t)(y1 * 80 + x0) * 256 + c];
  float d11 = base[(size_t)(y1 * 80 + x1) * 256 + c];
  float v = d00 * (1.f - wx) * (1.f - wy) + d01 * wx * (1.f - wy) +
            d10 * (1.f - wx) * wy + d11 * wx * wy;
  __shared__ float red[256];
  red[c] = v * v;
  __syncthreads();
  for (int s = 128; s > 0; s >>= 1) {
    if (c < s) red[c] += red[c + s];
    __syncthreads();
  }
  float nrm = fmaxf(sqrtf(red[0]), 1e-12f);
  desc_out[((size_t)b * 256 + c) * 1024 + t] = v / nrm;
}

// ---------------------------------------------------------------------------
// host driver
// ---------------------------------------------------------------------------
extern "C" void kernel_launch(void* const* d_in, const int* in_sizes, int n_in,
                              void* d_out, int out_size, void* d_ws,
                              size_t ws_size, hipStream_t stream) {
  (void)in_sizes; (void)n_in; (void)out_size; (void)ws_size;
  const float* w1a = (const float*)d_in[0];  const float* b1a = (const float*)d_in[1];
  const float* w1b = (const float*)d_in[2];  const float* b1b = (const float*)d_in[3];
  const float* w2a = (const float*)d_in[4];  const float* b2a = (const float*)d_in[5];
  const float* w2b = (const float*)d_in[6];  const float* b2b = (const float*)d_in[7];
  const float* w3a = (const float*)d_in[8];  const float* b3a = (const float*)d_in[9];
  const float* w3b = (const float*)d_in[10]; const float* b3b = (const float*)d_in[11];
  const float* w4a = (const float*)d_in[12]; const float* b4a = (const float*)d_in[13];
  const float* w4b = (const float*)d_in[14]; const float* b4b = (const float*)d_in[15];
  const float* wPa = (const float*)d_in[16]; const float* bPa = (const float*)d_in[17];
  const float* wPb = (const float*)d_in[18]; const float* bPb = (const float*)d_in[19];
  const float* wDa = (const float*)d_in[20]; const float* bDa = (const float*)d_in[21];
  const float* wDb = (const float*)d_in[22]; const float* bDb = (const float*)d_in[23];
  const float* img = (const float*)d_in[24];
  float* dout = (float*)d_out;

  char* base = (char*)d_ws;
  size_t off = 0;
  auto carve = [&](size_t bytes) -> void* {
    void* p = base + off;
    off += (bytes + 255) & ~(size_t)255;
    return p;
  };

  _Float16* actA = (_Float16*)carve((size_t)78643200 * 2);   // 4*480*640*64 fp16
  _Float16* actB = (_Float16*)carve((size_t)78643200 * 2);
  float* scoresRaw = (float*)carve((size_t)4 * 4800 * 80 * 4);
  float* descRaw   = (float*)carve((size_t)4 * 4800 * 256 * 4);
  const size_t NS = (size_t)4 * 480 * 640;
  float* s0  = (float*)carve(NS * 4);
  float* phb = (float*)carve(NS * 4);
  float* pp  = (float*)carve(NS * 4);
  float* mm  = (float*)carve(NS * 4);
  float* ssb = (float*)carve(NS * 4);
  float* p2  = (float*)carve(NS * 4);
  float* cScore = (float*)carve((size_t)4 * 8192 * 4);
  int*   cIdx   = (int*)carve((size_t)4 * 8192 * 4);
  int*   cnt    = (int*)carve(256);
  _Float16* pk1b = (_Float16*)carve((size_t)36864 * 2);
  _Float16* pk2a = (_Float16*)carve((size_t)36864 * 2);
  _Float16* pk2b = (_Float16*)carve((size_t)36864 * 2);
  _Float16* pk3a = (_Float16*)carve((size_t)73728 * 2);
  _Float16* pk3b = (_Float16*)carve((size_t)147456 * 2);
  _Float16* pk4a = (_Float16*)carve((size_t)147456 * 2);
  _Float16* pk4b = (_Float16*)carve((size_t)147456 * 2);
  _Float16* pkPa = (_Float16*)carve((size_t)294912 * 2);
  _Float16* pkPb = (_Float16*)carve((size_t)20480 * 2);
  _Float16* pkDa = (_Float16*)carve((size_t)294912 * 2);
  _Float16* pkDb = (_Float16*)carve((size_t)65536 * 2);

  auto pack = [&](const float* w, _Float16* dst, int CIN, int KHW, int ct,
                  int NK, int Oreal) {
    int total = ct * NK * 512;
    k_pack_w<<<(total + 255) / 256, 256, 0, stream>>>(w, dst, CIN, KHW, NK,
                                                      total, Oreal);
  };
  pack(w1b, pk1b, 64, 9, 4, 18, 64);
  pack(w2a, pk2a, 64, 9, 4, 18, 64);
  pack(w2b, pk2b, 64, 9, 4, 18, 64);
  pack(w3a, pk3a, 64, 9, 8, 18, 128);
  pack(w3b, pk3b, 128, 9, 8, 36, 128);
  pack(w4a, pk4a, 128, 9, 8, 36, 128);
  pack(w4b, pk4b, 128, 9, 8, 36, 128);
  pack(wPa, pkPa, 128, 9, 16, 36, 256);
  pack(wPb, pkPb, 256, 1, 5, 8, 65);
  pack(wDa, pkDa, 128, 9, 16, 36, 256);
  pack(wDb, pkDb, 256, 1, 16, 8, 256);

  // backbone
  k_conv1a<<<307200, 256, 0, stream>>>(img, w1a, b1a, actA);
  k_conv3x3_wmma<64, 64, 2, true><<<dim3(20, 480, 4), 256, 0, stream>>>(
      actA, pk1b, b1b, actB, 480, 640);
  k_maxpool2<<<76800, 256, 0, stream>>>(actB, actA, 240, 320, 64);
  k_conv3x3_wmma<64, 64, 2, true><<<dim3(10, 240, 4), 256, 0, stream>>>(
      actA, pk2a, b2a, actB, 240, 320);
  k_conv3x3_wmma<64, 64, 2, true><<<dim3(10, 240, 4), 256, 0, stream>>>(
      actB, pk2b, b2b, actA, 240, 320);
  k_maxpool2<<<19200, 256, 0, stream>>>(actA, actB, 120, 160, 64);
  k_conv3x3_wmma<64, 128, 2, true><<<dim3(5, 120, 8), 256, 0, stream>>>(
      actB, pk3a, b3a, actA, 120, 160);
  k_conv3x3_wmma<128, 128, 2, true><<<dim3(5, 120, 8), 256, 0, stream>>>(
      actA, pk3b, b3b, actB, 120, 160);
  k_maxpool2<<<9600, 256, 0, stream>>>(actB, actA, 60, 80, 128);
  k_conv3x3_wmma<128, 128, 1, true><<<dim3(5, 60, 4), 256, 0, stream>>>(
      actA, pk4a, b4a, actB, 60, 80);
  k_conv3x3_wmma<128, 128, 1, true><<<dim3(5, 60, 4), 256, 0, stream>>>(
      actB, pk4b, b4b, actA, 60, 80);

  // detector head
  k_conv3x3_wmma<128, 256, 1, true><<<dim3(5, 60, 8), 256, 0, stream>>>(
      actA, pkPa, bPa, actB, 60, 80);
  k_conv1x1_wmma<256, 80, false><<<dim3(300, 4, 1), 256, 0, stream>>>(
      actB, pkPb, bPb, 65, scoresRaw, 4800);

  // descriptor head (reuses actB after convPb consumed it)
  k_conv3x3_wmma<128, 256, 1, true><<<dim3(5, 60, 8), 256, 0, stream>>>(
      actA, pkDa, bDa, actB, 60, 80);
  k_conv1x1_wmma<256, 256, false><<<dim3(300, 4, 2), 256, 0, stream>>>(
      actB, pkDb, bDb, 256, descRaw, 4800);

  // softmax + pixel shuffle + simple_nms
  k_softmax_shuffle<<<75, 256, 0, stream>>>(scoresRaw, s0);
  const int EW = 4800;  // blocks for 4*480*640 elementwise
  k_pool9h<<<EW, 256, 0, stream>>>(s0, phb);
  k_pool9v<<<EW, 256, 0, stream>>>(phb, pp);
  k_eqmask<<<EW, 256, 0, stream>>>(s0, pp, mm);
  for (int it = 0; it < 2; ++it) {
    k_pool9h<<<EW, 256, 0, stream>>>(mm, phb);
    k_pool9v<<<EW, 256, 0, stream>>>(phb, pp);        // pooled mask
    k_suppress<<<EW, 256, 0, stream>>>(s0, pp, ssb);  // supp scores
    k_pool9h<<<EW, 256, 0, stream>>>(ssb, phb);
    k_pool9v<<<EW, 256, 0, stream>>>(phb, p2);        // pooled supp scores
    k_update<<<EW, 256, 0, stream>>>(mm, ssb, p2, pp);
  }

  // top-k
  k_zero4<<<1, 4, 0, stream>>>(cnt);
  k_compact<<<EW, 256, 0, stream>>>(s0, mm, cScore, cIdx, cnt);
  k_topk_sort<<<4, 1024, 0, stream>>>(cScore, cIdx, cnt, dout);

  // descriptors
  k_desc_norm<<<19200, 256, 0, stream>>>(descRaw);
  k_desc_sample<<<4096, 256, 0, stream>>>(descRaw, dout, dout + 12288);
}
